// Prunus_28613072126865
// MI455X (gfx1250) — matrix operand order repaired and checked by reading, not compiled, not run on hardware
//
#include <hip/hip_runtime.h>
#include <hip/hip_bf16.h>
#include <stdint.h>
#include <stddef.h>

// ---------------- types ----------------
typedef __bf16 bf16_t;
typedef __attribute__((ext_vector_type(16))) bf16_t v16bf;
typedef __attribute__((ext_vector_type(8)))  float  v8f;
typedef __attribute__((ext_vector_type(4)))  unsigned int uvec4;
typedef __attribute__((ext_vector_type(4)))  float  fvec4;

union BFrag { uvec4 u[2]; v16bf v; };          // 32B operand fragment (8 VGPRs)
union Pack8 { unsigned short s[8]; uvec4 u; }; // 8 packed bf16

// problem sizes (fixed by the reference)
#define BROWS 32768
#define DDIM  2048
#define HDIM  1024
#define LDIM  384
#define PDIM  192
#define NCAT  768   // [W_disc(384) ; W1_flat(384)]
#define CDIM  10

__device__ __forceinline__ unsigned short f2bf(float f) {
  unsigned int u = __float_as_uint(f);
  unsigned int r = 0x7FFFu + ((u >> 16) & 1u);   // round-to-nearest-even
  return (unsigned short)((u + r) >> 16);
}

// ---------------- CDNA5 async copy helpers ----------------
__device__ __forceinline__ void wait_async0() {
#if __has_builtin(__builtin_amdgcn_s_wait_asynccnt)
  __builtin_amdgcn_s_wait_asynccnt(0);
#else
  asm volatile("s_wait_asynccnt 0x0" ::: "memory");
#endif
}

// global_load_async_to_lds_b128: VDST = LDS byte address (low 32 bits of the
// generic shared pointer == LDS offset per ISA aperture rules), VADDR = u32
// byte offset, SADDR = 64-bit base. Tracked by ASYNCcnt.
__device__ __forceinline__ void async_ld_b128(unsigned ldsAddr,
                                              unsigned long long gbase,
                                              unsigned gByteOff) {
  asm volatile("global_load_async_to_lds_b128 %0, %1, %2"
               :: "v"(ldsAddr), "v"(gByteOff), "s"(gbase)
               : "memory");
}

// ---------------- fp32 -> bf16 convert (8 elems/thread) ----------------
__global__ void cvt_f32_bf16_kernel(const float* __restrict__ in,
                                    unsigned short* __restrict__ out, size_t n8) {
  size_t i = (size_t)blockIdx.x * blockDim.x + threadIdx.x;
  if (i >= n8) return;
  const fvec4* p = (const fvec4*)in + i * 2;
  fvec4 a = p[0], b = p[1];
  Pack8 o;
  o.s[0] = f2bf(a.x); o.s[1] = f2bf(a.y); o.s[2] = f2bf(a.z); o.s[3] = f2bf(a.w);
  o.s[4] = f2bf(b.x); o.s[5] = f2bf(b.y); o.s[6] = f2bf(b.z); o.s[7] = f2bf(b.w);
  ((uvec4*)out)[i] = o.u;
}

// ---------------- bf16 WMMA GEMM:  C[M,N] = A[M,K] * W[N,K]^T + bias[N] ----------------
// WG tile 128x128, 8 waves, each wave 32(M) x 64(N) = 2x4 WMMA tiles, Ktile=64.
// Double-buffered LDS fed by global_load_async_to_lds_b128 (ASYNCcnt).
#define TKT 64
#define LDT 72                 // padded LDS row stride (bf16); 144B, 16B-aligned
#define TILE_BYTES (128 * LDT * 2)   // 18432 B per A or B tile
#define SMEM_BYTES (4 * TILE_BYTES)  // A0,B0,A1,B1 = 73728 B
#define LDC 132                // padded fp32 epilogue row stride (528 B)

__global__ __launch_bounds__(256)
void gemm_bf16_kernel(const unsigned short* __restrict__ A,   // [M,K] bf16
                      const unsigned short* __restrict__ W,   // [N,K] bf16
                      const float* __restrict__ bias,         // [N]
                      float* __restrict__ C,                  // [M,N]
                      int M, int N, int K) {
  __shared__ __align__(16) char smem[SMEM_BYTES];
  unsigned short* lds = (unsigned short*)smem;
  const unsigned ldsBase = (unsigned)(uintptr_t)&smem[0];

  const int tid    = threadIdx.x;
  const int lane   = tid & 31;
  const int wave   = tid >> 5;
  const int wm     = wave & 3;      // M sub-tile 0..3  (32 rows each)
  const int wn     = wave >> 2;     // N sub-tile 0..1  (64 cols each)
  const int mblk   = blockIdx.x * 128;
  const int nblk   = blockIdx.y * 128;
  const int lane16 = lane & 15;
  const int laneHi = lane >> 4;     // 0|1

  const unsigned long long baseA = (unsigned long long)(uintptr_t)A;
  const unsigned long long baseW = (unsigned long long)(uintptr_t)W;

  // per-thread staging coordinates: 4 b128 per tile (1024 uint4 / 256 threads)
  // uid = tid + 256*c : row = uid>>3, col8 = (uid&7)*8
  v8f acc[2][4];
#pragma unroll
  for (int i = 0; i < 2; ++i)
#pragma unroll
    for (int j = 0; j < 4; ++j)
      acc[i][j] = (v8f){0.f,0.f,0.f,0.f,0.f,0.f,0.f,0.f};

  // ---- async stage of one (A,B) tile pair into buffer `bsel` ----
  auto stage = [&](int k0, int bsel) {
    unsigned bufOff = (unsigned)bsel * (2u * TILE_BYTES);
#pragma unroll
    for (int c = 0; c < 4; ++c) {
      int uid  = tid + 256 * c;
      int row  = uid >> 3;
      int col8 = (uid & 7) * 8;
      unsigned lOff = bufOff + (unsigned)(row * LDT + col8) * 2u;
      unsigned gA = (unsigned)(((unsigned)(mblk + row) * (unsigned)K +
                                (unsigned)(k0 + col8)) * 2u);
      unsigned gW = (unsigned)(((unsigned)(nblk + row) * (unsigned)K +
                                (unsigned)(k0 + col8)) * 2u);
      async_ld_b128(ldsBase + lOff, baseA, gA);
      async_ld_b128(ldsBase + lOff + TILE_BYTES, baseW, gW);
    }
  };

  stage(0, 0);
  int buf = 0;
  for (int k0 = 0; k0 < K; k0 += TKT) {
    wait_async0();       // own async loads landed
    __syncthreads();     // everyone's landed; prev buffer fully consumed
    if (k0 + TKT < K) stage(k0 + TKT, buf ^ 1);   // overlap next tile with compute

    const unsigned short* lA = lds + (buf ? 2 * (TILE_BYTES / 2) * 2 : 0) / 1;
    // (explicit): element offsets
    const unsigned short* pA = (const unsigned short*)(smem + buf * 2 * TILE_BYTES);
    const unsigned short* pB = (const unsigned short*)(smem + buf * 2 * TILE_BYTES + TILE_BYTES);
    (void)lA;

#pragma unroll
    for (int kk = 0; kk < TKT; kk += 32) {
      BFrag af[2], bfr[4];
      // A fragment (ISA 16-bit A 16x32): lane halves hold K {0..7,16..23}/{8..15,24..31}
#pragma unroll
      for (int tm = 0; tm < 2; ++tm) {
        int row = wm * 32 + tm * 16 + lane16;
        int kb  = kk + laneHi * 8;
        af[tm].u[0] = *(const uvec4*)&pA[row * LDT + kb];        // K = kb..kb+7
        af[tm].u[1] = *(const uvec4*)&pA[row * LDT + kb + 16];   // K = kb+16..kb+23
      }
      // B fragment (K x N): lane halves hold K {0..7,16..23}/{8..15,24..31}; N = lane&15
#pragma unroll
      for (int tn = 0; tn < 4; ++tn) {
        int row = wn * 64 + tn * 16 + lane16;
        int kb  = kk + laneHi * 8;
        bfr[tn].u[0] = *(const uvec4*)&pB[row * LDT + kb];
        bfr[tn].u[1] = *(const uvec4*)&pB[row * LDT + kb + 16];
      }
#pragma unroll
      for (int tm = 0; tm < 2; ++tm)
#pragma unroll
        for (int tn = 0; tn < 4; ++tn)
          acc[tm][tn] = __builtin_amdgcn_wmma_f32_16x16x32_bf16(
              false, af[tm].v, false, bfr[tn].v,
              (short)0, acc[tm][tn], false, false);
    }
    buf ^= 1;
  }

  // ---- epilogue: stage C tile through LDS for fully coalesced b128 stores ----
  __syncthreads();                       // all waves done reading bf16 tiles
  float* ldc = (float*)smem;             // 128 x LDC fp32 (67584 B <= 73728 B)
#pragma unroll
  for (int tn = 0; tn < 4; ++tn) {
    int nloc = wn * 64 + tn * 16 + lane16;
    float b  = bias[nblk + nloc];
#pragma unroll
    for (int tm = 0; tm < 2; ++tm) {
      int mloc = wm * 32 + tm * 16 + laneHi * 8;
#pragma unroll
      for (int r = 0; r < 8; ++r)
        ldc[(mloc + r) * LDC + nloc] = acc[tm][tn][r] + b;
    }
  }
  __syncthreads();
  // each warp streams whole 512B rows: uid = tid + 256*c; row = uid/32, col4 = (uid%32)*4
#pragma unroll
  for (int c = 0; c < 16; ++c) {
    int uid  = tid + 256 * c;
    int row  = uid >> 5;
    int col4 = (uid & 31) * 4;
    fvec4 v = *(const fvec4*)&ldc[row * LDC + col4];
    *(fvec4*)&C[(size_t)(mblk + row) * N + nblk + col4] = v;
  }
}

// ---------------- batch-norm statistics (sum / sumsq per column) ----------------
__global__ void bn_stats_kernel(const float* __restrict__ Z, float* __restrict__ sums,
                                int R, int Ccols, int ld, int rowsPerBlk) {
  int col = blockIdx.x * blockDim.x + threadIdx.x;
  if (col >= Ccols) return;
  int r0 = blockIdx.y * rowsPerBlk;
  int r1 = r0 + rowsPerBlk; if (r1 > R) r1 = R;
  float s = 0.f, s2 = 0.f;
  for (int r = r0; r < r1; ++r) {
    float v = Z[(size_t)r * ld + col];
    s += v; s2 += v * v;
  }
  atomicAdd(&sums[col], s);
  atomicAdd(&sums[Ccols + col], s2);
}

__global__ void bn_finalize_kernel(const float* __restrict__ sums,
                                   const float* __restrict__ g, const float* __restrict__ be,
                                   float* __restrict__ scale, float* __restrict__ shift,
                                   int Ccols, float invR) {
  int c = blockIdx.x * blockDim.x + threadIdx.x;
  if (c >= Ccols) return;
  float mu  = sums[c] * invR;
  float var = sums[Ccols + c] * invR - mu * mu;
  float sc  = g[c] * rsqrtf(var + 1e-5f);
  scale[c] = sc;
  shift[c] = be[c] - mu * sc;
}

// ---------------- BN + ReLU + pack to bf16 (8 elems/thread) ----------------
__global__ void bn_relu_bf16_kernel(const float* __restrict__ Y,
                                    const float* __restrict__ scale,
                                    const float* __restrict__ shift,
                                    unsigned short* __restrict__ out,
                                    int Ccols, size_t n8) {
  size_t i = (size_t)blockIdx.x * blockDim.x + threadIdx.x;
  if (i >= n8) return;
  size_t e   = i * 8;
  int    c0  = (int)(e % (size_t)Ccols);
  const fvec4* p = (const fvec4*)(Y + e);
  fvec4 a = p[0], b = p[1];
  Pack8 o;
  float v;
  v = fmaxf(0.f, a.x * scale[c0+0] + shift[c0+0]); o.s[0] = f2bf(v);
  v = fmaxf(0.f, a.y * scale[c0+1] + shift[c0+1]); o.s[1] = f2bf(v);
  v = fmaxf(0.f, a.z * scale[c0+2] + shift[c0+2]); o.s[2] = f2bf(v);
  v = fmaxf(0.f, a.w * scale[c0+3] + shift[c0+3]); o.s[3] = f2bf(v);
  v = fmaxf(0.f, b.x * scale[c0+4] + shift[c0+4]); o.s[4] = f2bf(v);
  v = fmaxf(0.f, b.y * scale[c0+5] + shift[c0+5]); o.s[5] = f2bf(v);
  v = fmaxf(0.f, b.z * scale[c0+6] + shift[c0+6]); o.s[6] = f2bf(v);
  v = fmaxf(0.f, b.w * scale[c0+7] + shift[c0+7]); o.s[7] = f2bf(v);
  ((uvec4*)out)[i] = o.u;
}

// ---------------- fused tail: disc BN+ReLU, heads, gumbel route, expert L2 ----------------
__device__ __forceinline__ float waveSum(float v) {
#pragma unroll
  for (int off = 16; off > 0; off >>= 1) v += __shfl_xor(v, off, 32);
  return v;
}

__global__ __launch_bounds__(256)
void tail_kernel(const float* __restrict__ Z, int ldz,           // [B, 768]
                 const float* __restrict__ scaleD, const float* __restrict__ shiftD,
                 const float* __restrict__ Wdfc,  const float* __restrict__ bdfc,
                 const float* __restrict__ Wsw,   const float* __restrict__ bsw,
                 const float* __restrict__ gnoise,
                 const float* __restrict__ W2,    const float* __restrict__ b2,
                 float* __restrict__ cop, float* __restrict__ dom,
                 int* __restrict__ idxOut, int B) {
  int lane = threadIdx.x & 31;
  int row  = blockIdx.x * (blockDim.x >> 5) + (threadIdx.x >> 5);
  if (row >= B) return;
  const float* z = Z + (size_t)row * ldz;

  // dp = relu(BN(z[0:384])); heads: dom (2), sw (2)
  float pd0 = 0.f, pd1 = 0.f, ps0 = 0.f, ps1 = 0.f;
#pragma unroll
  for (int i = 0; i < 12; ++i) {
    int j = lane + 32 * i;
    float dp = fmaxf(0.f, z[j] * scaleD[j] + shiftD[j]);
    pd0 += dp * Wdfc[j];
    pd1 += dp * Wdfc[LDIM + j];
    ps0 += dp * Wsw[j];
    ps1 += dp * Wsw[LDIM + j];
  }
  pd0 = waveSum(pd0); pd1 = waveSum(pd1);
  ps0 = waveSum(ps0); ps1 = waveSum(ps1);

  float dom0 = pd0 + bdfc[0], dom1 = pd1 + bdfc[1];
  float sw0  = fmaxf(0.f, ps0 + bsw[0]);
  float sw1  = fmaxf(0.f, ps1 + bsw[1]);
  float l0   = (sw0 + gnoise[(size_t)row * 2 + 0]) * 10.0f;  // /TAU, TAU=0.1
  float l1   = (sw1 + gnoise[(size_t)row * 2 + 1]) * 10.0f;
  float m    = fmaxf(l0, l1);
  float e0   = __expf(l0 - m), e1 = __expf(l1 - m);
  float inv  = 1.0f / (e0 + e1);
  float sumg = (e0 + e1) * inv;          // == sum(softmax) (≈1), faithful to reference
  int   idx  = (l1 > l0) ? 1 : 0;        // argmax, first-max on tie

  // selected expert second layer: h = relu(z[384+idx*192 : +192]);  o = h @ W2[idx].T + b2
  const float* zh = z + LDIM + idx * PDIM;
  float o[CDIM];
#pragma unroll
  for (int c = 0; c < CDIM; ++c) o[c] = 0.f;
#pragma unroll
  for (int i = 0; i < 6; ++i) {
    int p = lane + 32 * i;
    float h = fmaxf(0.f, zh[p]);
#pragma unroll
    for (int c = 0; c < CDIM; ++c)
      o[c] += h * W2[((size_t)idx * CDIM + c) * PDIM + p];
  }
#pragma unroll
  for (int c = 0; c < CDIM; ++c) o[c] = waveSum(o[c]);

  if (lane == 0) {
    dom[(size_t)row * 2 + 0] = dom0;
    dom[(size_t)row * 2 + 1] = dom1;
    idxOut[row] = idx;
#pragma unroll
    for (int c = 0; c < CDIM; ++c)
      cop[(size_t)row * CDIM + c] = (o[c] + b2[idx * CDIM + c]) * sumg;
  }
}

// ---------------- host orchestration ----------------
extern "C" void kernel_launch(void* const* d_in, const int* in_sizes, int n_in,
                              void* d_out, int out_size, void* d_ws, size_t ws_size,
                              hipStream_t stream) {
  const float* x      = (const float*)d_in[0];
  const float* W_pre  = (const float*)d_in[1];
  const float* b_pre  = (const float*)d_in[2];
  const float* g_pre  = (const float*)d_in[3];
  const float* be_pre = (const float*)d_in[4];
  const float* W_disc = (const float*)d_in[5];
  const float* b_disc = (const float*)d_in[6];
  const float* g_disc = (const float*)d_in[7];
  const float* be_disc= (const float*)d_in[8];
  const float* W_dfc  = (const float*)d_in[9];
  const float* b_dfc  = (const float*)d_in[10];
  const float* W_sw   = (const float*)d_in[11];
  const float* b_sw   = (const float*)d_in[12];
  const float* W1     = (const float*)d_in[13];
  const float* b1     = (const float*)d_in[14];
  const float* W2     = (const float*)d_in[15];
  const float* b2     = (const float*)d_in[16];
  const float* gnoise = (const float*)d_in[17];

  char* ws = (char*)d_ws;
  // region 0: x_bf16 (128MB) -> later reused as feature_bf16 (64MB)
  unsigned short* xbf    = (unsigned short*)(ws);
  unsigned short* featbf = (unsigned short*)(ws);
  // region 1: Y1 fp32 (128MB) -> later reused as Z fp32 (96MB)
  float* Y1 = (float*)(ws + (size_t)134217728);
  float* Z  = Y1;
  // region 2: small aux
  char* aux = ws + (size_t)268435456;
  unsigned short* wprebf = (unsigned short*)(aux);                      // 4 MB
  unsigned short* wcatbf = (unsigned short*)(aux + 4194304);            // 1.5 MB
  float* bcat   = (float*)(aux + 4194304 + 1572864);                    // 3 KB
  float* sums1  = bcat + NCAT;                                          // 2*1024
  float* scale1 = sums1 + 2 * HDIM;
  float* shift1 = scale1 + HDIM;
  float* sums2  = shift1 + HDIM;                                        // 2*384
  float* scale2 = sums2 + 2 * LDIM;
  float* shift2 = scale2 + LDIM;

  float* copO = (float*)d_out;                       // [32768,10]
  float* domO = copO + (size_t)BROWS * CDIM;         // [32768,2]
  int*   idxO = (int*)d_out + (size_t)BROWS * (CDIM + 2);  // [32768] int32

  // zero atomic accumulators (deterministic each call)
  hipMemsetAsync(sums1, 0, 2 * HDIM * sizeof(float), stream);
  hipMemsetAsync(sums2, 0, 2 * LDIM * sizeof(float), stream);

  // converts: x, W_pre, and Wcat = [W_disc ; W1_flat] ; bcat = [b_disc ; b1_flat]
  {
    size_t n8 = (size_t)BROWS * DDIM / 8;
    cvt_f32_bf16_kernel<<<(unsigned)((n8 + 255) / 256), 256, 0, stream>>>(x, xbf, n8);
  }
  {
    size_t n8 = (size_t)HDIM * DDIM / 8;
    cvt_f32_bf16_kernel<<<(unsigned)((n8 + 255) / 256), 256, 0, stream>>>(W_pre, wprebf, n8);
  }
  {
    size_t n8 = (size_t)LDIM * HDIM / 8;
    cvt_f32_bf16_kernel<<<(unsigned)((n8 + 255) / 256), 256, 0, stream>>>(W_disc, wcatbf, n8);
    cvt_f32_bf16_kernel<<<(unsigned)((n8 + 255) / 256), 256, 0, stream>>>(
        W1, wcatbf + (size_t)LDIM * HDIM, n8);
  }
  hipMemcpyAsync(bcat,        b_disc, LDIM * sizeof(float), hipMemcpyDeviceToDevice, stream);
  hipMemcpyAsync(bcat + LDIM, b1,     LDIM * sizeof(float), hipMemcpyDeviceToDevice, stream);

  // GEMM1: Y1 = x @ W_pre^T + b_pre     [32768,2048]x[2048,1024]
  {
    dim3 grid(BROWS / 128, HDIM / 128);
    gemm_bf16_kernel<<<grid, 256, 0, stream>>>(xbf, wprebf, b_pre, Y1,
                                               BROWS, HDIM, DDIM);
  }
  // BN stats + finalize for pre layer
  {
    dim3 grid((HDIM + 255) / 256, 64);
    bn_stats_kernel<<<grid, 256, 0, stream>>>(Y1, sums1, BROWS, HDIM, HDIM, BROWS / 64);
    bn_finalize_kernel<<<(HDIM + 255) / 256, 256, 0, stream>>>(
        sums1, g_pre, be_pre, scale1, shift1, HDIM, 1.0f / (float)BROWS);
  }
  // feature = bf16(relu(BN(Y1)))  (overwrites x_bf16 region)
  {
    size_t n8 = (size_t)BROWS * HDIM / 8;
    bn_relu_bf16_kernel<<<(unsigned)((n8 + 255) / 256), 256, 0, stream>>>(
        Y1, scale1, shift1, featbf, HDIM, n8);
  }
  // GEMM2: Z = feature @ [W_disc;W1]^T + bcat   [32768,1024]x[1024,768] (overwrites Y1)
  {
    dim3 grid(BROWS / 128, NCAT / 128);
    gemm_bf16_kernel<<<grid, 256, 0, stream>>>(featbf, wcatbf, bcat, Z,
                                               BROWS, NCAT, HDIM);
  }
  // BN stats + finalize for disc layer (first 384 cols of Z, ld=768)
  {
    dim3 grid((LDIM + 255) / 256, 64);
    bn_stats_kernel<<<grid, 256, 0, stream>>>(Z, sums2, BROWS, LDIM, NCAT, BROWS / 64);
    bn_finalize_kernel<<<(LDIM + 255) / 256, 256, 0, stream>>>(
        sums2, g_disc, be_disc, scale2, shift2, LDIM, 1.0f / (float)BROWS);
  }
  // fused tail: one wave per row
  {
    dim3 grid(BROWS / 8);
    tail_kernel<<<grid, 256, 0, stream>>>(Z, NCAT, scale2, shift2,
                                          W_dfc, b_dfc, W_sw, b_sw, gnoise,
                                          W2, b2, copO, domO, idxO, BROWS);
  }
}
